// NeuralInterestNetwork_13503377179003
// MI455X (gfx1250) — matrix operand order repaired and checked by reading, not compiled, not run on hardware
//
#include <hip/hip_runtime.h>
#include <hip/hip_bf16.h>

// MI455X (gfx1250, wave32). Edge-scoring GNN:
//   out[e] = lrelu(lrelu(c1[src]+c2[dst]) @ Wmlp^T + b) @ w_hL
//          + (x[src]*m[dst]) @ w_h1 + (w[src]*m[dst]) @ w_h2 + biases
// Node tables (~102 MB) fit in the 192 MB L2 -> gather-bound; matmul pushed
// onto v_wmma_f32_16x16x32_f16 (16 edges per wave = one M-tile, K=N=64).

typedef __attribute__((ext_vector_type(16))) _Float16 v16h;
typedef __attribute__((ext_vector_type(8)))  float    v8f;

#define DD 64
#define NEG_SLOPE 0.01f

__device__ __forceinline__ float lrelu(float v) { return v >= 0.0f ? v : NEG_SLOPE * v; }
__device__ __forceinline__ float4 ld4(const float* p) { return *(const float4*)p; }

// ---- precompute c = a + b over node tables (float4 vectorized) ----
__global__ __launch_bounds__(256)
void node_sum_kernel(const float* __restrict__ a, const float* __restrict__ b,
                     float* __restrict__ o, int n4) {
    int i = blockIdx.x * blockDim.x + threadIdx.x;
    if (i < n4) {
        float4 av = ld4(a + 4l * i);
        float4 bv = ld4(b + 4l * i);
        float4 ov = make_float4(av.x + bv.x, av.y + bv.y, av.z + bv.z, av.w + bv.w);
        *(float4*)(o + 4l * i) = ov;
    }
}

// ---- pre-pack Wmlp^T into WMMA B-fragment lane layout (f16) ----
// Fragment (kc,nt), lane = half*16+n': elem i = Wmlp[nt*16+n'][kc*32 + 16*half + i]
__global__ __launch_bounds__(256)
void pack_b_kernel(const float* __restrict__ Wmlp, _Float16* __restrict__ Bw) {
    const int t    = threadIdx.x;        // 0..255 = (kc*4+nt)*32 + lane
    const int lane = t & 31;
    const int frag = t >> 5;
    const int kc   = frag >> 2;
    const int nt   = frag & 3;
    const int n    = nt * 16 + (lane & 15);
    const int kb   = kc * 32 + 16 * (lane >> 4);
#pragma unroll
    for (int i = 0; i < 16; ++i)
        Bw[t * 16 + i] = (_Float16)Wmlp[n * DD + kb + i];
}

// ---- main edge kernel: 8 waves/block, 16 edges per wave ----
template <bool FUSE_NODES, bool PRE_B>
__global__ __launch_bounds__(256)
void edge_kernel(const float* __restrict__ s,  const float* __restrict__ p,
                 const float* __restrict__ x,  const float* __restrict__ w,
                 const float* __restrict__ q1, const float* __restrict__ q2,
                 const float* __restrict__ m,
                 const int* __restrict__ src,  const int* __restrict__ dst,
                 const float* __restrict__ Wmlp, const float* __restrict__ bmlp,
                 const float* __restrict__ whL,  const float* __restrict__ bhL,
                 const float* __restrict__ wh1,  const float* __restrict__ bh1,
                 const float* __restrict__ wh2,  const float* __restrict__ bh2,
                 const float* __restrict__ c1,   const float* __restrict__ c2,
                 const _Float16* __restrict__ Bw,
                 float* __restrict__ out, int n_edges)
{
    __shared__ float s_mlp[8][16][17];   // [wave][edge][writer-lane%16], padded
    __shared__ float s_ew [8][16][2];    // [wave][edge][half]

    const int lane = threadIdx.x & 31;
    const int wv   = threadIdx.x >> 5;
    const int r    = lane & 15;          // A row / B column residue / edge-in-tile
    const int half = lane >> 4;

    const int e0 = (blockIdx.x * 8 + wv) * 16;
    int e = e0 + r;
    if (e >= n_edges) e = n_edges - 1;   // clamp; WMMA needs EXEC all-ones
    const long si = (long)src[e] * DD;
    const long di = (long)dst[e] * DD;

    // ================= WMMA MLP: H = lrelu(C) x Wmlp^T =================
    v8f acc[4] = {};                     // 4 N-tiles of 16 cols, f32 accum
#pragma unroll
    for (int kc = 0; kc < 2; ++kc) {     // K chunks of 32
        // A fragment (contiguous order, matches B): elem i(0..15) -> k = kc*32+16h+i
        float fa[16];
#pragma unroll
        for (int seg = 0; seg < 2; ++seg) {
            const int k = kc * 32 + 16 * half + seg * 8;
#pragma unroll
            for (int q = 0; q < 2; ++q) {
                float4 u, v;
                if (FUSE_NODES) {
                    float4 a0 = ld4(s  + si + k + q * 4);
                    float4 a1 = ld4(p  + si + k + q * 4);
                    float4 b0 = ld4(q1 + di + k + q * 4);
                    float4 b1 = ld4(q2 + di + k + q * 4);
                    u = make_float4(a0.x + a1.x, a0.y + a1.y, a0.z + a1.z, a0.w + a1.w);
                    v = make_float4(b0.x + b1.x, b0.y + b1.y, b0.z + b1.z, b0.w + b1.w);
                } else {
                    u = ld4(c1 + si + k + q * 4);
                    v = ld4(c2 + di + k + q * 4);
                }
                const int o = seg * 8 + q * 4;
                fa[o + 0] = lrelu(u.x + v.x);
                fa[o + 1] = lrelu(u.y + v.y);
                fa[o + 2] = lrelu(u.z + v.z);
                fa[o + 3] = lrelu(u.w + v.w);
            }
        }
        v16h av;
#pragma unroll
        for (int i = 0; i < 16; ++i) av[i] = (_Float16)fa[i];

#pragma unroll
        for (int nt = 0; nt < 4; ++nt) {
            v16h bv;
            if (PRE_B) {
                bv = *(const v16h*)(Bw + (((kc * 4 + nt) * 32) + lane) * 16);
            } else {
                const int n  = nt * 16 + r;
                const int kb = kc * 32 + 16 * half;
#pragma unroll
                for (int q = 0; q < 4; ++q) {
                    float4 wv4 = ld4(Wmlp + n * DD + kb + q * 4);
                    bv[q * 4 + 0] = (_Float16)wv4.x;
                    bv[q * 4 + 1] = (_Float16)wv4.y;
                    bv[q * 4 + 2] = (_Float16)wv4.z;
                    bv[q * 4 + 3] = (_Float16)wv4.w;
                }
            }
            acc[nt] = __builtin_amdgcn_wmma_f32_16x16x32_f16(
                false, av, false, bv, (short)0, acc[nt], false, false);
        }
    }

    // ======= MLP head: lane holds H[8*half+g][nt*16+r], g=0..7 =======
    float pm[8];
#pragma unroll
    for (int g = 0; g < 8; ++g) pm[g] = 0.0f;
#pragma unroll
    for (int nt = 0; nt < 4; ++nt) {
        const int n = nt * 16 + r;
        const float bn = bmlp[n];
        const float wn = whL[n];
#pragma unroll
        for (int g = 0; g < 8; ++g)
            pm[g] += lrelu(acc[nt][g] + bn) * wn;
    }
#pragma unroll
    for (int g = 0; g < 8; ++g)
        s_mlp[wv][8 * half + g][r] = pm[g];

    // ======= elementwise heads: m*(x*w_h1 + w*w_h2), split over 2 lanes =======
    float ew = 0.0f;
#pragma unroll
    for (int kc = 0; kc < 2; ++kc) {
#pragma unroll
        for (int seg = 0; seg < 2; ++seg) {
            const int k = kc * 32 + seg * 16 + 8 * half;
#pragma unroll
            for (int q = 0; q < 2; ++q) {
                float4 xm = ld4(x   + si + k + q * 4);
                float4 wm = ld4(w   + si + k + q * 4);
                float4 mm = ld4(m   + di + k + q * 4);
                float4 h1 = ld4(wh1 + k + q * 4);
                float4 h2 = ld4(wh2 + k + q * 4);
                ew += mm.x * (xm.x * h1.x + wm.x * h2.x);
                ew += mm.y * (xm.y * h1.y + wm.y * h2.y);
                ew += mm.z * (xm.z * h1.z + wm.z * h2.z);
                ew += mm.w * (xm.w * h1.w + wm.w * h2.w);
            }
        }
    }
    s_ew[wv][r][half] = ew;

    __syncthreads();

    if (lane < 16) {
        const int eo = e0 + lane;
        if (eo < n_edges) {
            float o = bhL[0] + bh1[0] + bh2[0]
                    + s_ew[wv][lane][0] + s_ew[wv][lane][1];
#pragma unroll
            for (int j = 0; j < 16; ++j) o += s_mlp[wv][lane][j];
            out[eo] = o;
        }
    }
}

extern "C" void kernel_launch(void* const* d_in, const int* in_sizes, int n_in,
                              void* d_out, int out_size, void* d_ws, size_t ws_size,
                              hipStream_t stream) {
    (void)n_in; (void)out_size;
    const float* s    = (const float*)d_in[0];
    const float* p    = (const float*)d_in[1];
    const float* x    = (const float*)d_in[2];
    const float* w    = (const float*)d_in[3];
    const float* q1   = (const float*)d_in[4];
    const float* q2   = (const float*)d_in[5];
    const float* m    = (const float*)d_in[6];
    const int*   src  = (const int*)d_in[7];
    const int*   dst  = (const int*)d_in[8];
    const float* Wmlp = (const float*)d_in[9];
    const float* bmlp = (const float*)d_in[10];
    const float* whL  = (const float*)d_in[11];
    const float* bhL  = (const float*)d_in[12];
    const float* wh1  = (const float*)d_in[13];
    const float* bh1  = (const float*)d_in[14];
    const float* wh2  = (const float*)d_in[15];
    const float* bh2  = (const float*)d_in[16];
    float* out = (float*)d_out;

    const int  E  = in_sizes[7];
    const long UD = in_sizes[0];   // N_USERS * 64
    const long ID = in_sizes[4];   // N_ITEMS * 64

    const size_t BW_BYTES = (size_t)2 * 4 * 32 * 16 * sizeof(_Float16); // 8 KB
    const size_t c1B = (size_t)UD * sizeof(float);
    const size_t c2B = (size_t)ID * sizeof(float);

    char* wsb = (char*)d_ws;
    _Float16* Bw = (_Float16*)wsb;
    float* c1 = (float*)(wsb + BW_BYTES);
    float* c2 = (float*)(wsb + BW_BYTES + c1B);

    const bool preB = ws_size >= BW_BYTES;
    const bool preN = ws_size >= BW_BYTES + c1B + c2B;

    if (preB) pack_b_kernel<<<1, 256, 0, stream>>>(Wmlp, Bw);
    if (preN) {
        node_sum_kernel<<<(int)((UD / 4 + 255) / 256), 256, 0, stream>>>(s, p, c1, (int)(UD / 4));
        node_sum_kernel<<<(int)((ID / 4 + 255) / 256), 256, 0, stream>>>(q1, q2, c2, (int)(ID / 4));
    }

    const int blocks = (E + 127) / 128;  // 128 edges per block (8 waves x 16)
    if (preN) {
        edge_kernel<false, true><<<blocks, 256, 0, stream>>>(
            s, p, x, w, q1, q2, m, src, dst, Wmlp, bmlp, whL, bhL, wh1, bh1, wh2, bh2,
            c1, c2, Bw, out, E);
    } else if (preB) {
        edge_kernel<true, true><<<blocks, 256, 0, stream>>>(
            s, p, x, w, q1, q2, m, src, dst, Wmlp, bmlp, whL, bhL, wh1, bh1, wh2, bh2,
            nullptr, nullptr, Bw, out, E);
    } else {
        edge_kernel<true, false><<<blocks, 256, 0, stream>>>(
            s, p, x, w, q1, q2, m, src, dst, Wmlp, bmlp, whL, bhL, wh1, bh1, wh2, bh2,
            nullptr, nullptr, nullptr, out, E);
    }
}
